// LayerStacks_45234595561653
// MI455X (gfx1250) — compile-verified, matrix operand order checked by reading, not run on hardware
//
#include <hip/hip_runtime.h>
#include <stdint.h>

// LayerStacks bucketed MLP for MI455X (gfx1250, wave32).
// Memory-bound: x traffic 128MB -> ~5.5us floor @ 23.3TB/s. Layer-1 GEMM done
// with v_wmma_f32_16x16x32_bf16 (f32 accum) against fused Weff = W1 + tile(W1f),
// all 64 bucket-features computed (shared B matrix), bucket gathered in epilogue.
// f32->bf16 conversion: hardware v_cvt_pk_bf16_f32 (RNE) via inline asm.
// Weff staging: global_load_async_to_lds_b128 (ASYNCcnt) -- no VGPR round-trip.

#define KDIM   2048   // 2*L1
#define L2D    8
#define L3D    32
#define NBUCK  8
#define NF     64     // NBUCK * L2D = total layer-1 features
#define KC     128    // K chunk staged in LDS per iteration

typedef __attribute__((ext_vector_type(16))) __bf16 v16bf;
typedef __attribute__((ext_vector_type(8)))  float  v8f;

// Hardware packed convert: dst[15:0] = bf16(lo), dst[31:16] = bf16(hi), RNE.
__device__ __forceinline__ uint32_t cvt_pk_bf16(float lo, float hi) {
  uint32_t d;
  asm("v_cvt_pk_bf16_f32 %0, %1, %2" : "=v"(d) : "v"(lo), "v"(hi));
  return d;
}

// Build Weff[row][k] = W1[row][k] + W1f[row%8][k], stored as packed bf16 pairs
// (uint32 per 2 elements), row-major, row stride KDIM/2 uints.
__global__ void ls_prep_weff(const float* __restrict__ W1,
                             const float* __restrict__ W1f,
                             uint32_t* __restrict__ weff) {
  int t = blockIdx.x * blockDim.x + threadIdx.x;      // over NF * KDIM/2 pairs
  if (t >= NF * (KDIM / 2)) return;
  int row = t / (KDIM / 2);
  int cp  = t % (KDIM / 2);
  const float* w1r = W1  + (size_t)row * KDIM + cp * 2;
  const float* wfr = W1f + (size_t)(row & (L2D - 1)) * KDIM + cp * 2;
  weff[t] = cvt_pk_bf16(w1r[0] + wfr[0], w1r[1] + wfr[1]);
}

__launch_bounds__(256)
__global__ void ls_main(const float* __restrict__ x,
                        const int* __restrict__ lsidx,
                        const uint32_t* __restrict__ weff,
                        const float* __restrict__ b1,
                        const float* __restrict__ W2,
                        const float* __restrict__ b2,
                        const float* __restrict__ Wout,
                        const float* __restrict__ bout,
                        float* __restrict__ out) {
  __shared__ uint4 sW[NF * KC / 8];          // 16KB: staged Weff K-chunk (bf16)
  __shared__ float sD[8][16][NF];            // 32KB: per-wave C tiles
  __shared__ float sB1[NF];
  __shared__ float sW2[NBUCK * L3D * L2D];   // 8KB
  __shared__ float sB2[NBUCK * L3D];
  __shared__ float sWo[NBUCK * L3D];
  __shared__ float sBo[NBUCK];

  const int tid  = threadIdx.x;
  const int wave = tid >> 5;
  const int lane = tid & 31;
  const int r    = lane & 15;    // row-within-tile / column index
  const int h    = lane >> 4;    // lane half selects K sub-range
  const int s0   = blockIdx.x * 128 + wave * 16;   // this wave's first sample

  // Preload small parameters into LDS.
  for (int i = tid; i < NF; i += 256) sB1[i] = b1[i];
  for (int i = tid; i < NBUCK * L3D * L2D; i += 256) sW2[i] = W2[i];
  for (int i = tid; i < NBUCK * L3D; i += 256) { sB2[i] = b2[i]; sWo[i] = Wout[i]; }
  if (tid < NBUCK) sBo[tid] = bout[tid];

  // Raw LDS byte address of sW (generic LDS pointers: offset lives in addr[31:0]).
  const uint32_t sWlds = (uint32_t)(uintptr_t)&sW[0];

  const float* xrow = x + (size_t)(s0 + r) * KDIM;

  v8f acc0 = {}, acc1 = {}, acc2 = {}, acc3 = {};

  for (int k0 = 0; k0 < KDIM; k0 += KC) {
    __syncthreads();   // previous chunk fully consumed before restage
    // Async stage of Weff[:, k0:k0+KC] -> LDS (1024 b128 transfers / 16KB).
    // GVS mode: SGPR base = weff, per-lane 32-bit byte offset, tracked by ASYNCcnt.
    for (int i = tid; i < NF * (KC / 8); i += 256) {
      int row = i >> 4;       // 16 b128 transfers per row-chunk
      int off = i & 15;
      uint32_t goff  = (uint32_t)row * (KDIM * 2) + (uint32_t)k0 * 2 + (uint32_t)off * 16;
      uint32_t laddr = sWlds + (uint32_t)i * 16;
      asm volatile("global_load_async_to_lds_b128 %0, %1, %2"
                   :: "v"(laddr), "v"(goff), "s"(weff) : "memory");
    }
    // Speculative prefetch of next x chunk (global_prefetch_b8).
    if (k0 + KC < KDIM)
      __builtin_prefetch(xrow + k0 + KC + h * 8, 0, 1);
    asm volatile("s_wait_asynccnt 0x0" ::: "memory");   // our async ops done
    __syncthreads();                                    // all waves' staging visible

    for (int kk = 0; kk < KC; kk += 32) {
      // --- A fragment: 16x32 bf16 of x-tile ---
      // lane h=0: K = kk+{0..7, 16..23}; lane h=1: K = kk+{8..15, 24..31}
      union { v16bf v; uint32_t w[8]; } A;
      const float* xp = xrow + k0 + kk + h * 8;
      float4 f0 = *(const float4*)(xp);
      float4 f1 = *(const float4*)(xp + 4);
      float4 f2 = *(const float4*)(xp + 16);
      float4 f3 = *(const float4*)(xp + 20);
      A.w[0] = cvt_pk_bf16(f0.x, f0.y); A.w[1] = cvt_pk_bf16(f0.z, f0.w);
      A.w[2] = cvt_pk_bf16(f1.x, f1.y); A.w[3] = cvt_pk_bf16(f1.z, f1.w);
      A.w[4] = cvt_pk_bf16(f2.x, f2.y); A.w[5] = cvt_pk_bf16(f2.z, f2.w);
      A.w[6] = cvt_pk_bf16(f3.x, f3.y); A.w[7] = cvt_pk_bf16(f3.z, f3.w);

      // --- B fragments: lane n holds Weff row (nt*16 + n), same K pattern ---
#pragma unroll
      for (int nt = 0; nt < 4; ++nt) {
        union { v16bf v; uint4 q[2]; } B;
        const uint4* bp = &sW[(nt * 16 + r) * (KC / 8) + ((kk + h * 8) >> 3)];
        B.q[0] = bp[0];     // K offsets +0..7   (of this lane's sub-range)
        B.q[1] = bp[2];     // K offsets +16..23 (+16 bf16 = +2 uint4)
        v8f& acc = (nt == 0 ? acc0 : nt == 1 ? acc1 : nt == 2 ? acc2 : acc3);
        acc = __builtin_amdgcn_wmma_f32_16x16x32_bf16(
            false, A.v, false, B.v, (short)0, acc, false, false);
      }
    }
  }

  // Spill C tiles: VGPR j, lanes 0-15 -> (M=j, N=lane); lanes 16-31 -> (M=j+8).
#pragma unroll
  for (int j = 0; j < 8; ++j) {
    int m = j + h * 8;
    sD[wave][m][ 0 + r] = acc0[j];
    sD[wave][m][16 + r] = acc1[j];
    sD[wave][m][32 + r] = acc2[j];
    sD[wave][m][48 + r] = acc3[j];
  }
  __syncthreads();

  // Epilogue: lanes 0-15 each finish one sample (bucket gather + layers 2/3).
  if (lane < 16) {
    const int m   = r;
    const int s   = s0 + m;
    const int bkt = lsidx[s];
    float l1v[L2D];
#pragma unroll
    for (int j = 0; j < L2D; ++j) {
      float v = sD[wave][m][bkt * L2D + j] + sB1[bkt * L2D + j];
      l1v[j] = fminf(fmaxf(v, 0.0f), 1.0f);
    }
    float o = sBo[bkt];
#pragma unroll
    for (int n = 0; n < L3D; ++n) {
      const float* w2r = &sW2[(bkt * L3D + n) * L2D];
      float a = sB2[bkt * L3D + n];
#pragma unroll
      for (int j = 0; j < L2D; ++j) a += w2r[j] * l1v[j];
      a = fminf(fmaxf(a, 0.0f), 1.0f);
      o += sWo[bkt * L3D + n] * a;
    }
    out[s] = o;
  }
}

extern "C" void kernel_launch(void* const* d_in, const int* in_sizes, int n_in,
                              void* d_out, int out_size, void* d_ws, size_t ws_size,
                              hipStream_t stream) {
  const float* x    = (const float*)d_in[0];
  const int*   idx  = (const int*)d_in[1];
  const float* W1   = (const float*)d_in[2];
  const float* b1   = (const float*)d_in[3];
  const float* W1f  = (const float*)d_in[4];
  const float* W2   = (const float*)d_in[5];
  const float* b2   = (const float*)d_in[6];
  const float* Wout = (const float*)d_in[7];
  const float* bout = (const float*)d_in[8];
  float* out = (float*)d_out;
  uint32_t* weff = (uint32_t*)d_ws;   // NF * KDIM/2 uints = 256KB

  const int nB = in_sizes[0] / KDIM;  // 16384 samples

  dim3 pgrid((NF * (KDIM / 2) + 255) / 256);
  ls_prep_weff<<<pgrid, 256, 0, stream>>>(W1, W1f, weff);

  dim3 grid(nB / 128);                // 8 waves x 16 samples per workgroup
  ls_main<<<grid, 256, 0, stream>>>(x, idx, weff, b1, W2, b2, Wout, bout, out);
}
